// GAM_RHN_3633542333097
// MI455X (gfx1250) — matrix operand order repaired and checked by reading, not compiled, not run on hardware
//
#include <hip/hip_runtime.h>
#include <hip/hip_bf16.h>
#include <math.h>

// ---------------- problem dims ----------------
#define EMBED   512
#define HIDDEN  1024
#define CLASSES 10
#define BATCH   64
#define SEQ     512
#define ROWS    (BATCH * SEQ)          // 32768

typedef __bf16 bf16_t;
typedef __bf16 v16bf __attribute__((ext_vector_type(16)));
typedef __bf16 v8bf  __attribute__((ext_vector_type(8)));
typedef float  v8f   __attribute__((ext_vector_type(8)));

// ---------------- WMMA helpers (CDNA5 16x16x32 bf16 -> f32) ----------------
// A layout (16x32, MxK, 16-bit): lane 0-15 row=lane holds K[0..7]+K[16..23];
// lane 16-31 row=lane-16 holds K[8..15]+K[24..31]  (pairs packed per VGPR).
__device__ __forceinline__ v16bf load_a_tile(const bf16_t* __restrict__ A, int lda,
                                             int row0, int kbase, int lane) {
    const int row  = row0 + (lane & 15);
    const int koff = (lane >> 4) * 8;
    const bf16_t* p = A + (size_t)row * lda + kbase + koff;
    v8bf lo = *(const v8bf*)(p);
    v8bf hi = *(const v8bf*)(p + 16);
    v16bf a;
#pragma unroll
    for (int i = 0; i < 8; ++i) { a[i] = lo[i]; a[8 + i] = hi[i]; }
    return a;
}

// B layout (32x16, KxN, 16-bit): lane 0-15 col=lane holds K[0..7]+K[16..23];
// lane 16-31 col=lane-16 holds K[8..15]+K[24..31].  BT is B transposed: BT[n][k].
__device__ __forceinline__ v16bf load_b_tile(const bf16_t* __restrict__ BT, int ldb,
                                             int col0, int kbase, int lane) {
    const int col  = col0 + (lane & 15);
    const int koff = (lane >> 4) * 8;
    const bf16_t* p = BT + (size_t)col * ldb + kbase + koff;
    v8bf lo = *(const v8bf*)(p);
    v8bf hi = *(const v8bf*)(p + 16);
    v16bf b;
#pragma unroll
    for (int i = 0; i < 8; ++i) { b[i] = lo[i]; b[8 + i] = hi[i]; }
    return b;
}

__device__ __forceinline__ v8f wmma_bf16(v16bf a, v16bf b, v8f c) {
    return __builtin_amdgcn_wmma_f32_16x16x32_bf16(
        /*neg_a=*/false, a, /*neg_b=*/false, b,
        /*c_mod=*/(short)0, c, /*reuse_a=*/false, /*reuse_b=*/false);
}

// ---------------- prep kernels ----------------
// Gather embedding rows per token and convert to bf16: Aemb[r][e] = emb[x[r]][e]
__global__ __launch_bounds__(256) void gather_emb_kernel(const int* __restrict__ x,
                                                         const float* __restrict__ emb,
                                                         bf16_t* __restrict__ Aemb) {
    const size_t idx = (size_t)blockIdx.x * blockDim.x + threadIdx.x;
    if (idx >= (size_t)ROWS * EMBED) return;
    const int r = (int)(idx >> 9);        // EMBED == 512
    const int e = (int)(idx & 511);
    const int tok = x[r];
    Aemb[idx] = (bf16_t)emb[(size_t)tok * EMBED + e];
}

// dst[n][k] = (bf16) src[k][n]   (src is K x N row-major)
__global__ __launch_bounds__(256) void transpose_bf16_kernel(const float* __restrict__ src,
                                                             bf16_t* __restrict__ dst,
                                                             int K, int N) {
    const size_t idx = (size_t)blockIdx.x * blockDim.x + threadIdx.x;
    if (idx >= (size_t)K * N) return;
    const int n = (int)(idx / K);
    const int k = (int)(idx % K);
    dst[idx] = (bf16_t)src[(size_t)k * N + n];
}

// ---------------- input projection: X[r][h] = Aemb[r][:] @ W[:, h] + bias[h] ----------------
// One 16x16 tile per wave; 4 waves per block (consecutive n-tiles share A rows).
__global__ __launch_bounds__(128) void proj_kernel(const bf16_t* __restrict__ Aemb,   // [ROWS][EMBED]
                                                   const bf16_t* __restrict__ WT,     // [HIDDEN][EMBED]
                                                   const float*  __restrict__ bias,   // [HIDDEN]
                                                   bf16_t* __restrict__ X) {          // [ROWS][HIDDEN]
    const int lane = threadIdx.x & 31;
    const int gw   = blockIdx.x * 4 + (threadIdx.x >> 5);
    const int m_tile = gw >> 6;         // 0..2047
    const int n_tile = gw & 63;         // 0..63
    v8f acc = {};
#pragma unroll 4
    for (int k = 0; k < EMBED; k += 32) {
        v16bf a = load_a_tile(Aemb, EMBED, m_tile * 16, k, lane);
        v16bf b = load_b_tile(WT,   EMBED, n_tile * 16, k, lane);
        acc = wmma_bf16(a, b, acc);
    }
    const int col   = n_tile * 16 + (lane & 15);
    const int rbase = m_tile * 16 + ((lane >> 4) * 8);
    const float bv  = bias[col];
#pragma unroll
    for (int v = 0; v < 8; ++v) {
        const int r = rbase + v;
        X[(size_t)r * HIDDEN + col] = (bf16_t)(acc[v] + bv);
    }
}

// ---------------- one recurrence step ----------------
// s = tanh(xh_t + h @ R_h); t = sigmoid(xt_t + h @ R_t); h' = s*t + h*(1-t)
__global__ __launch_bounds__(128) void step_kernel(const bf16_t* __restrict__ hb_prev, // [B][H] bf16
                                                   const float*  __restrict__ hf_prev, // [B][H] f32
                                                   const bf16_t* __restrict__ RhT,     // [H][H]
                                                   const bf16_t* __restrict__ RtT,     // [H][H]
                                                   const bf16_t* __restrict__ XH,      // [ROWS][H]
                                                   const bf16_t* __restrict__ XT,      // [ROWS][H]
                                                   int t,
                                                   float*  __restrict__ hf_next,
                                                   bf16_t* __restrict__ hb_next) {
    const int lane = threadIdx.x & 31;
    const int gw   = blockIdx.x * 4 + (threadIdx.x >> 5);
    const int m_tile = gw >> 6;         // 0..3   (batch tiles)
    const int n_tile = gw & 63;         // 0..63  (hidden tiles)
    v8f acc_s = {};
    v8f acc_t = {};
#pragma unroll 2
    for (int k = 0; k < HIDDEN; k += 32) {
        v16bf a  = load_a_tile(hb_prev, HIDDEN, m_tile * 16, k, lane);
        v16bf b1 = load_b_tile(RhT,     HIDDEN, n_tile * 16, k, lane);
        acc_s = wmma_bf16(a, b1, acc_s);
        v16bf b2 = load_b_tile(RtT,     HIDDEN, n_tile * 16, k, lane);
        acc_t = wmma_bf16(a, b2, acc_t);
    }
    const int col   = n_tile * 16 + (lane & 15);
    const int rbase = m_tile * 16 + ((lane >> 4) * 8);
#pragma unroll
    for (int v = 0; v < 8; ++v) {
        const int b = rbase + v;                          // batch index
        const size_t xoff = ((size_t)b * SEQ + t) * HIDDEN + col;
        const float xh = (float)XH[xoff];
        const float xt = (float)XT[xoff];
        const float hp = hf_prev[(size_t)b * HIDDEN + col];
        const float sg = tanhf(acc_s[v] + xh);
        const float zg = acc_t[v] + xt;
        const float tg = 1.0f / (1.0f + __expf(-zg));
        const float hn = sg * tg + hp * (1.0f - tg);
        hf_next[(size_t)b * HIDDEN + col] = hn;
        hb_next[(size_t)b * HIDDEN + col] = (bf16_t)hn;
    }
}

// ---------------- final FC: out[b][c] = h[b][:] @ W_fc[:,c] + b_fc[c] ----------------
__global__ __launch_bounds__(256) void fc_kernel(const float* __restrict__ h,
                                                 const float* __restrict__ Wfc,
                                                 const float* __restrict__ bfc,
                                                 float* __restrict__ out) {
    const int idx = blockIdx.x * blockDim.x + threadIdx.x;
    if (idx >= BATCH * CLASSES) return;
    const int b = idx / CLASSES;
    const int c = idx % CLASSES;
    float acc = bfc[c];
    for (int k = 0; k < HIDDEN; ++k)
        acc += h[(size_t)b * HIDDEN + k] * Wfc[(size_t)k * CLASSES + c];
    out[idx] = acc;
}

// ---------------- host launcher ----------------
extern "C" void kernel_launch(void* const* d_in, const int* in_sizes, int n_in,
                              void* d_out, int out_size, void* d_ws, size_t ws_size,
                              hipStream_t stream) {
    const int*   x    = (const int*)  d_in[0];
    const float* emb  = (const float*)d_in[1];
    const float* W_h  = (const float*)d_in[2];
    const float* R_h  = (const float*)d_in[3];
    const float* b_h  = (const float*)d_in[4];
    const float* W_t  = (const float*)d_in[5];
    const float* R_t  = (const float*)d_in[6];
    const float* b_t  = (const float*)d_in[7];
    const float* W_fc = (const float*)d_in[8];
    const float* b_fc = (const float*)d_in[9];
    float* out = (float*)d_out;

    char* ws = (char*)d_ws;
    size_t off = 0;
    auto salloc = [&](size_t bytes) -> void* {
        void* p = ws + off;
        off += (bytes + 255) & ~(size_t)255;
        return p;
    };

    bf16_t* Aemb = (bf16_t*)salloc((size_t)ROWS * EMBED * 2);        // 33.5 MB
    bf16_t* WhT  = (bf16_t*)salloc((size_t)HIDDEN * EMBED * 2);      // 1 MB
    bf16_t* WtT  = (bf16_t*)salloc((size_t)HIDDEN * EMBED * 2);      // 1 MB
    bf16_t* RhT  = (bf16_t*)salloc((size_t)HIDDEN * HIDDEN * 2);     // 2 MB
    bf16_t* RtT  = (bf16_t*)salloc((size_t)HIDDEN * HIDDEN * 2);     // 2 MB
    bf16_t* XH   = (bf16_t*)salloc((size_t)ROWS * HIDDEN * 2);       // 64 MB
    bf16_t* XT   = (bf16_t*)salloc((size_t)ROWS * HIDDEN * 2);       // 64 MB
    float*  hf0  = (float*) salloc((size_t)BATCH * HIDDEN * 4);
    float*  hf1  = (float*) salloc((size_t)BATCH * HIDDEN * 4);
    bf16_t* hb0  = (bf16_t*)salloc((size_t)BATCH * HIDDEN * 2);
    bf16_t* hb1  = (bf16_t*)salloc((size_t)BATCH * HIDDEN * 2);

    // --- prep: gather/convert embedding, transpose/convert weights ---
    {
        const size_t n = (size_t)ROWS * EMBED;
        gather_emb_kernel<<<(unsigned)((n + 255) / 256), 256, 0, stream>>>(x, emb, Aemb);
    }
    {
        const size_t nw = (size_t)EMBED * HIDDEN;
        transpose_bf16_kernel<<<(unsigned)((nw + 255) / 256), 256, 0, stream>>>(W_h, WhT, EMBED, HIDDEN);
        transpose_bf16_kernel<<<(unsigned)((nw + 255) / 256), 256, 0, stream>>>(W_t, WtT, EMBED, HIDDEN);
        const size_t nr = (size_t)HIDDEN * HIDDEN;
        transpose_bf16_kernel<<<(unsigned)((nr + 255) / 256), 256, 0, stream>>>(R_h, RhT, HIDDEN, HIDDEN);
        transpose_bf16_kernel<<<(unsigned)((nr + 255) / 256), 256, 0, stream>>>(R_t, RtT, HIDDEN, HIDDEN);
    }

    // --- input projections (WMMA GEMM): XH/XT = Aemb @ W + bias ---
    {
        const int tiles  = (ROWS / 16) * (HIDDEN / 16);   // 2048 * 64 = 131072 wave-tiles
        const int blocks = tiles / 4;                     // 4 waves / block
        proj_kernel<<<blocks, 128, 0, stream>>>(Aemb, WhT, b_h, XH);
        proj_kernel<<<blocks, 128, 0, stream>>>(Aemb, WtT, b_t, XT);
    }

    // --- h0 = 0 ---
    hipMemsetAsync(hf0, 0, (size_t)BATCH * HIDDEN * 4, stream);
    hipMemsetAsync(hb0, 0, (size_t)BATCH * HIDDEN * 2, stream);

    // --- 512 sequential recurrence steps (ping-pong) ---
    float*  hf[2] = {hf0, hf1};
    bf16_t* hb[2] = {hb0, hb1};
    int cur = 0;
    for (int t = 0; t < SEQ; ++t) {
        const int nxt = cur ^ 1;
        // 256 tiles, 1 wave each, 4 waves per block -> 64 blocks
        step_kernel<<<64, 128, 0, stream>>>(hb[cur], hf[cur], RhT, RtT, XH, XT, t,
                                            hf[nxt], hb[nxt]);
        cur = nxt;
    }

    // --- final FC ---
    fc_kernel<<<(BATCH * CLASSES + 255) / 256, 256, 0, stream>>>(hf[cur], W_fc, b_fc, out);
}
